// TTLinear_12232066859472
// MI455X (gfx1250) — matrix operand order repaired and checked by reading, not compiled, not run on hardware
//
#include <hip/hip_runtime.h>

// ---------------------------------------------------------------------------
// TT-linear (in_modes 16^3 -> out_modes 16^3, ranks 1,16,16,1, B=2048), fused.
// bf16 WMMA (v_wmma_f32_16x16x32_bf16), f32 accumulation. All weights are
// pre-baked into WMMA B-fragment order in d_ws, then async-copied into LDS
// once per workgroup (global_load_async_to_lds_b128 / s_wait_asynccnt), so
// the hot loop touches only LDS + the x row. One wave per batch row.
// ---------------------------------------------------------------------------

typedef __attribute__((ext_vector_type(16))) __bf16 v16bf;
typedef __attribute__((ext_vector_type(8)))  __bf16 v8bf;
typedef __attribute__((ext_vector_type(8)))  float  v8f;

#define B_ROWS       2048
#define WAVES_PER_WG 8

// d_ws / LDS fragment block layout (unsigned short elements):
//   [0      , 65536) : w1 B-fragments, 128 frags (j*8+kk) x 32 lanes x 16 bf16
//   [65536  , 73728) : w0 B-fragments, 16 frags (j)       x 32 lanes x 16 bf16 (elems 8-15 zero)
//   [73728  , 81920) : w2 B-fragments, 16 frags (m2)      x 32 lanes x 16 bf16 (elems 8-15 zero)
#define WS_W0F 65536
#define WS_W2F 73728
#define WS_TOTAL 81920

__device__ __forceinline__ unsigned short f2bf_u(float f) {
  // round-to-nearest-even f32 -> bf16 (bit pattern), scalar fallback
  unsigned u = __builtin_bit_cast(unsigned, f);
  unsigned r = u + 0x7FFFu + ((u >> 16) & 1u);
  return (unsigned short)(r >> 16);
}

template <typename T>
__device__ __forceinline__ unsigned as_u32(T t) {
  if constexpr (sizeof(T) == 4) {
    return __builtin_bit_cast(unsigned, t);
  } else {
    unsigned u = 0;
    __builtin_memcpy(&u, &t, sizeof(T) < 4 ? sizeof(T) : 4);
    return u;
  }
}

// two f32 -> packed bf16 pair (single VALU op where the builtin exists)
__device__ __forceinline__ unsigned cvt_pk2(float a, float b) {
#if __has_builtin(__builtin_amdgcn_cvt_pk_bf16_f32)
  return as_u32(__builtin_amdgcn_cvt_pk_bf16_f32(a, b));
#else
  return (unsigned)f2bf_u(a) | ((unsigned)f2bf_u(b) << 16);
#endif
}

__device__ __forceinline__ void store_bf16_pair(unsigned short* p0, unsigned short* p1,
                                                float a, float b) {
  unsigned pk = cvt_pk2(a, b);
  *p0 = (unsigned short)(pk & 0xFFFFu);       // ds_store_b16
  *p1 = (unsigned short)(pk >> 16);           // ds_store_b16_d16_hi
}

// async global -> LDS copy, 16 bytes per lane, tracked on ASYNCcnt
__device__ __forceinline__ void async_copy_b128(unsigned lds_off, const void* gaddr) {
  asm volatile("global_load_async_to_lds_b128 %0, %1, off"
               :: "v"(lds_off), "v"(gaddr) : "memory");
}
__device__ __forceinline__ void wait_asynccnt0() {
  asm volatile("s_wait_asynccnt 0x0" ::: "memory");
}

union AB32 {                // one lane's share of a 16x32 bf16 WMMA operand
  v16bf          v;
  v8bf           h[2];
  unsigned       w[8];
  unsigned short u[16];
};

// ---------------------------------------------------------------------------
// Prep: bake all three weight matrices into bf16 WMMA B-fragment order.
// ---------------------------------------------------------------------------
__global__ void tt_prep_weights(const float* __restrict__ w0,
                                const float* __restrict__ w1,
                                const float* __restrict__ w2,
                                unsigned short* __restrict__ wsf) {
  int id = blockIdx.x * blockDim.x + threadIdx.x;   // 0..81919
  if (id < WS_W0F) {                                // ---- w1 fragments
    int f    = id >> 9;          // 0..127
    int r    = id & 511;
    int lane = r >> 4;
    int e    = r & 15;
    int n    = (f >> 3) * 16 + (lane & 15);
    int K    = (f & 7) * 32 + ((e >> 3) << 4) + ((lane >> 4) << 3) + (e & 7);
    wsf[id]  = f2bf_u(w1[n * 256 + K]);
  } else if (id < WS_W2F) {                         // ---- w0 fragments
    int q    = id - WS_W0F;
    int j    = q >> 9;
    int lane = (q >> 4) & 31;
    int e    = q & 15;
    wsf[id]  = (e < 8) ? f2bf_u(w0[(j * 16 + (lane & 15)) * 16 + (lane >> 4) * 8 + e]) : (unsigned short)0;
  } else if (id < WS_TOTAL) {                       // ---- w2 fragments
    int q    = id - WS_W2F;
    int m2   = q >> 9;
    int lane = (q >> 4) & 31;
    int e    = q & 15;
    wsf[id]  = (e < 8) ? f2bf_u(w2[(lane & 15) * 256 + m2 * 16 + (lane >> 4) * 8 + e]) : (unsigned short)0;
  }
}

// ---------------------------------------------------------------------------
// Main fused kernel: one wave (32 lanes) per batch row, 8 waves per WG.
// Dynamic LDS: 160 KB weight fragments. Static LDS: 8x16KB staging + pad.
// ---------------------------------------------------------------------------
extern __shared__ unsigned short s_wf[];    // WS_TOTAL bf16 fragments (160 KB)

__global__ __launch_bounds__(WAVES_PER_WG * 32)
void tt_linear_kernel(const float* __restrict__ x,
                      const float* __restrict__ bias,
                      const unsigned short* __restrict__ wsf,
                      float* __restrict__ out) {
  __shared__ unsigned short s_T0[WAVES_PER_WG][4096] __attribute__((aligned(16)));  // T0'[o0][K=(m1,r1)]
  __shared__ unsigned short s_T1[WAVES_PER_WG][4096] __attribute__((aligned(16)));  // T1 [o0][o1][r2]
  __shared__ unsigned short s_zpad[16] __attribute__((aligned(16)));                // zeroed 16B (+spare)

  const int tid = threadIdx.x;
  // ---- async-stage all weight fragments (160 KB) global -> LDS, no VGPRs
  for (int i = tid; i < WS_TOTAL / 8; i += WAVES_PER_WG * 32) {   // 16B chunks
    async_copy_b128((unsigned)(uintptr_t)(s_wf + i * 8), wsf + i * 8);
  }
  if (tid < 16) s_zpad[tid] = 0;
  wait_asynccnt0();
  __syncthreads();

  const int wave = tid >> 5;
  const int lane = tid & 31;
  const int lo   = lane & 15;           // tile row/col within half-wave
  const int hi8  = (lane >> 4) << 3;    // 0 for lanes 0-15, 8 for lanes 16-31
  const int b    = blockIdx.x * WAVES_PER_WG + wave;

  unsigned short* T0 = s_T0[wave];
  unsigned short* T1 = s_T1[wave];
  const unsigned short* w1f = s_wf;
  const unsigned short* w0f = s_wf + WS_W0F;
  const unsigned short* w2f = s_wf + WS_W2F;
  const v8bf zpad8 = *(const v8bf*)s_zpad;

  const float* xrow = x + (size_t)b * 4096;
  __builtin_prefetch(xrow, 0, 1);       // global_prefetch_b8 for the row

  const v8f vzero = {0.f, 0.f, 0.f, 0.f, 0.f, 0.f, 0.f, 0.f};
  v8f acc2[16];
#pragma unroll
  for (int t = 0; t < 16; ++t) acc2[t] = vzero;

  for (int m2 = 0; m2 < 16; ++m2) {
    // ---------------- step0: T0'[o0][(m1,r1)], 16 N-tiles, K=m0 (16, padded)
    AB32 a0;                            // A rows = m1 (=lo), K = m0
#pragma unroll
    for (int e = 0; e < 4; ++e) {
      float xa = xrow[(hi8 + 2 * e) * 256 + lo * 16 + m2];
      float xb = xrow[(hi8 + 2 * e + 1) * 256 + lo * 16 + m2];
      a0.w[e] = cvt_pk2(xa, xb);
    }
    a0.h[1] = zpad8;                    // K=16..31 zero pad

#pragma unroll 1
    for (int j = 0; j < 16; ++j) {      // N-tile j: n = 16j + r1, o0 = j
      AB32 b0;                          // pre-baked fragment (zeros included)
      const unsigned short* pb = w0f + j * 512 + lane * 16;
      b0.h[0] = *(const v8bf*)pb;
      b0.h[1] = *(const v8bf*)(pb + 8);
      v8f d = __builtin_amdgcn_wmma_f32_16x16x32_bf16(
          false, a0.v, false, b0.v, (short)0, vzero, false, false);
      // D (M=m1,N=r1) -> T0'[o0=j][m1*16+r1]
      unsigned short* q = T0 + j * 256 + hi8 * 16 + lo;
#pragma unroll
      for (int p8 = 0; p8 < 8; p8 += 2)
        store_bf16_pair(q + p8 * 16, q + (p8 + 1) * 16, d[p8], d[p8 + 1]);
    }

    // ---------------- step1: T1[o0][(o1,r2)] = T0' x w1^T, K=256 (all LDS)
#pragma unroll 1
    for (int j = 0; j < 16; ++j) {      // N-tile j: o1 = j
      v8f c1a = vzero;                  // two independent WMMA chains
      v8f c1b = vzero;
#pragma unroll
      for (int kk = 0; kk < 8; kk += 2) {
        AB32 a1, b1, a1x, b1x;
        const unsigned short* pa = T0 + lo * 256 + kk * 32 + hi8;
        const unsigned short* pb = w1f + (((j << 3) + kk) << 9) + lane * 16;
        a1.h[0]  = *(const v8bf*)pa;            // K = 32kk + hi8 + 0..7
        a1.h[1]  = *(const v8bf*)(pa + 16);     // K = 32kk + 16 + hi8 + 0..7
        b1.h[0]  = *(const v8bf*)pb;
        b1.h[1]  = *(const v8bf*)(pb + 8);
        a1x.h[0] = *(const v8bf*)(pa + 32);
        a1x.h[1] = *(const v8bf*)(pa + 48);
        b1x.h[0] = *(const v8bf*)(pb + 512);
        b1x.h[1] = *(const v8bf*)(pb + 520);
        c1a = __builtin_amdgcn_wmma_f32_16x16x32_bf16(
            false, a1.v, false, b1.v, (short)0, c1a, false, false);
        c1b = __builtin_amdgcn_wmma_f32_16x16x32_bf16(
            false, a1x.v, false, b1x.v, (short)0, c1b, false, false);
      }
      // D (M=o0,N=r2) -> T1[o0][o1=j][r2]
      unsigned short* q = T1 + hi8 * 256 + j * 16 + lo;
#pragma unroll
      for (int p8 = 0; p8 < 8; p8 += 2)
        store_bf16_pair(q + p8 * 256, q + (p8 + 1) * 256,
                        c1a[p8] + c1b[p8], c1a[p8 + 1] + c1b[p8 + 1]);
    }

    // ---------------- step2: acc2[t=o0] += T1 x w2^T-slice, K=r2 (16, padded)
    AB32 b2;                            // pre-baked fragment (zeros included)
    {
      const unsigned short* pb = w2f + m2 * 512 + lane * 16;
      b2.h[0] = *(const v8bf*)pb;
      b2.h[1] = *(const v8bf*)(pb + 8);
    }
#pragma unroll
    for (int t = 0; t < 16; ++t) {      // M-tile t: rows (o0=t, o1=0..15)
      AB32 a2;
      a2.h[0] = *(const v8bf*)(T1 + t * 256 + lo * 16 + hi8);  // K=r2=hi8+0..7
      a2.h[1] = *(const v8bf*)s_zpad;                          // K>=16 pad (LDS load)
      acc2[t] = __builtin_amdgcn_wmma_f32_16x16x32_bf16(
          false, a2.v, false, b2.v, (short)0, acc2[t], false, false);
    }
  }

  // ---------------- epilogue: out[b][(o0,o1,o2)] = acc + bias (coalesced b32)
  float* orow = out + (size_t)b * 4096;
#pragma unroll
  for (int t = 0; t < 16; ++t) {
#pragma unroll
    for (int p8 = 0; p8 < 8; ++p8) {
      int idx = t * 256 + (p8 + hi8) * 16 + lo;
      orow[idx] = acc2[t][p8] + bias[idx];
    }
  }
}

// ---------------------------------------------------------------------------
extern "C" void kernel_launch(void* const* d_in, const int* in_sizes, int n_in,
                              void* d_out, int out_size, void* d_ws, size_t ws_size,
                              hipStream_t stream) {
  const float* x    = (const float*)d_in[0];
  const float* w0   = (const float*)d_in[1];
  const float* w1   = (const float*)d_in[2];
  const float* w2   = (const float*)d_in[3];
  const float* bias = (const float*)d_in[4];

  unsigned short* wsf = (unsigned short*)d_ws;   // 81920 bf16 = 160 KB scratch

  tt_prep_weights<<<WS_TOTAL / 256, 256, 0, stream>>>(w0, w1, w2, wsf);
  // 160 KB dynamic LDS (weight fragments); CDNA5 allows up to 320 KB per WG.
  tt_linear_kernel<<<B_ROWS / WAVES_PER_WG, WAVES_PER_WG * 32,
                     WS_TOTAL * sizeof(unsigned short), stream>>>(
      x, bias, wsf, (float*)d_out);
}
